// Attention_sr_63960652972458
// MI455X (gfx1250) — matrix operand order repaired and checked by reading, not compiled, not run on hardware
//
#include <hip/hip_runtime.h>
#include <hip/hip_bf16.h>

// ---------------------------------------------------------------------------
// MI455X (gfx1250) multi-head attention: cvt -> QKV gemm -> flash attn -> proj
// All GEMMs use v_wmma_f32_16x16x32_bf16 (wave32). bf16 conversion is hoisted
// into a one-time elementwise pass so GEMM inner loops are loads + WMMA only.
// ---------------------------------------------------------------------------

#define BATCH   2
#define NTOK    2304      // 48*48
#define CDIM    768
#define NHEAD   12
#define HDIM    64
#define SCALEQ  0.125f    // 64^-0.5

typedef __attribute__((ext_vector_type(16))) __bf16 v16bf;
typedef __attribute__((ext_vector_type(8)))  __bf16 v8bf;
typedef __attribute__((ext_vector_type(8)))  float  v8f;

static __device__ __forceinline__ unsigned short bf_bits(float f) {
    unsigned u = __builtin_bit_cast(unsigned, f);
    unsigned r = 0x7FFFu + ((u >> 16) & 1u);       // round-to-nearest-even
    return (unsigned short)((u + r) >> 16);
}
static __device__ __forceinline__ __bf16 f2bf(float f) {
    unsigned short s = bf_bits(f);
    return __builtin_bit_cast(__bf16, s);
}
static __device__ __forceinline__ v16bf cat8(v8bf lo, v8bf hi) {
    return __builtin_shufflevector(lo, hi, 0,1,2,3,4,5,6,7,8,9,10,11,12,13,14,15);
}
static __device__ __forceinline__ v8f wmma_bf16(v16bf a, v16bf b, v8f c) {
    // D = A(16x32) * B(32x16) + C, f32 accumulate
    return __builtin_amdgcn_wmma_f32_16x16x32_bf16(false, a, false, b, (short)0, c,
                                                   false, false);
}

// ---------------------------------------------------------------------------
// Kernel 0: elementwise f32 -> bf16 (8 elements / lane), memory-bound.
// ---------------------------------------------------------------------------
__global__ __launch_bounds__(256)
void cvt_kernel(const float* __restrict__ src, __bf16* __restrict__ dst, int n8)
{
    const int i = blockIdx.x * 256 + threadIdx.x;
    if (i < n8) {
        v8f f = *(const v8f*)(src + (size_t)i * 8);
        v8bf o;
#pragma unroll
        for (int j = 0; j < 8; ++j) o[j] = f2bf(f[j]);
        *(v8bf*)(dst + (size_t)i * 8) = o;
    }
}

// ---------------------------------------------------------------------------
// Kernel 1: qkv = x @ qkv_w^T (both bf16); scatter into Q[b,h,n,d]*scale,
//           K[b,h,n,d], Vt[b,h,d,n] as bf16.
// Grid: (288 M-tiles, 36) x 128 threads; each wave owns one 16x16 tile,
// ft = blockIdx.y*4 + wave in [0,144)  (2304 output features / 16).
// ---------------------------------------------------------------------------
__global__ __launch_bounds__(128)
void qkv_kernel(const __bf16* __restrict__ x, const __bf16* __restrict__ w,
                __bf16* __restrict__ Qs, __bf16* __restrict__ Ks,
                __bf16* __restrict__ Vt)
{
    const int lane = threadIdx.x & 31;
    const int wave = threadIdx.x >> 5;
    const int l    = lane & 15;
    const int half = lane >> 4;
    const int mt   = blockIdx.x;
    const int ft   = blockIdx.y * 4 + wave;

    const __bf16* xrow = x + (size_t)(mt * 16 + l) * CDIM;
    const __bf16* wrow = w + (size_t)(ft * 16 + l) * CDIM;

    v8f acc = {};
    for (int k0 = 0; k0 < CDIM; k0 += 32) {
        // A fragment: lane<16 -> k 0..15, lane>=16 -> 16..31 (contiguous)
        v16bf a = *(const v16bf*)(xrow + k0 + half * 16);
        // B fragment: lane<16 -> k 0..15, lane>=16 -> 16..31 (contiguous)
        v16bf b = *(const v16bf*)(wrow + k0 + half * 16);
        acc = wmma_bf16(a, b, acc);
    }

    const int f_base = ft * 16;
    const int which  = f_base / CDIM;            // 0=q 1=k 2=v (constant per tile)
    const int head   = (f_base % CDIM) / HDIM;
    const int dd     = (f_base % HDIM) + l;      // output column for this lane
#pragma unroll
    for (int r = 0; r < 8; ++r) {
        const int m     = half * 8 + r;
        const int token = mt * 16 + m;
        const int bb    = token / NTOK;
        const int n     = token % NTOK;
        const float v   = acc[r];
        if (which == 0)
            Qs[((size_t)(bb * NHEAD + head) * NTOK + n) * HDIM + dd] = f2bf(v * SCALEQ);
        else if (which == 1)
            Ks[((size_t)(bb * NHEAD + head) * NTOK + n) * HDIM + dd] = f2bf(v);
        else
            Vt[((size_t)(bb * NHEAD + head) * HDIM + dd) * NTOK + n] = f2bf(v);
    }
}

// ---------------------------------------------------------------------------
// Kernel 2: flash attention per (b, head). Each wave owns a 16-query block,
// streams 32-key tiles: S = Q K^T (4 WMMAs over d=64), online softmax in
// registers, P re-laid-out via per-wave LDS tile, O += P Vt (4 WMMAs).
// Output -> AO bf16 in [b, n, h*d] layout for the projection GEMM.
// Grid: (36, 24) x 128 threads.
// ---------------------------------------------------------------------------
__global__ __launch_bounds__(128)
void attn_kernel(const __bf16* __restrict__ Qs, const __bf16* __restrict__ Ks,
                 const __bf16* __restrict__ Vt, __bf16* __restrict__ AO)
{
    __shared__ __align__(16) unsigned short ldsP[4][16][32];

    const int lane = threadIdx.x & 31;
    const int wave = threadIdx.x >> 5;
    const int l    = lane & 15;
    const int half = lane >> 4;
    const int rt   = blockIdx.x * 4 + wave;      // 16-query block index [0,144)
    const int bh   = blockIdx.y;
    const int bb   = bh / NHEAD;
    const int head = bh % NHEAD;

    const size_t hbase = (size_t)(bb * NHEAD + head);
    const __bf16* Qp = Qs + hbase * NTOK * HDIM;
    const __bf16* Kp = Ks + hbase * NTOK * HDIM;
    const __bf16* Vp = Vt + hbase * HDIM * NTOK; // [64][2304]

    // Q A-fragments for d 0..31 and 32..63 (held for the whole loop)
    const __bf16* qrow = Qp + (size_t)(rt * 16 + l) * HDIM;
    v16bf aq0 = *(const v16bf*)(qrow + half * 16);
    v16bf aq1 = *(const v16bf*)(qrow + 32 + half * 16);

    v8f oacc[4] = {v8f{}, v8f{}, v8f{}, v8f{}};
    float mi[8], li[8];
#pragma unroll
    for (int r = 0; r < 8; ++r) { mi[r] = -1e30f; li[r] = 0.0f; }

    for (int kt = 0; kt < NTOK; kt += 32) {
        // --- S = Q K^T for 32 keys (two 16-key column tiles) ---
        const __bf16* k0row = Kp + (size_t)(kt + l) * HDIM;
        const __bf16* k1row = Kp + (size_t)(kt + 16 + l) * HDIM;
        v16bf bk00 = *(const v16bf*)(k0row + half * 16);        // d 0..31
        v16bf bk01 = *(const v16bf*)(k0row + 32 + half * 16);   // d 32..63
        v16bf bk10 = *(const v16bf*)(k1row + half * 16);
        v16bf bk11 = *(const v16bf*)(k1row + 32 + half * 16);

        // prefetch next K tile (gfx1250 global_prefetch_b8 path)
        if (kt + 32 < NTOK)
            __builtin_prefetch((const void*)(Kp + (size_t)(kt + 32 + l) * HDIM), 0, 1);

        v8f z = {};
        v8f s0 = wmma_bf16(aq0, bk00, z);
        s0     = wmma_bf16(aq1, bk01, s0);
        v8f s1 = wmma_bf16(aq0, bk10, z);
        s1     = wmma_bf16(aq1, bk11, s1);

        // --- online softmax (rows live across the 16-lane half-wave) ---
#pragma unroll
        for (int r = 0; r < 8; ++r) {
            float t = fmaxf(s0[r], s1[r]);
            t = fmaxf(t, __shfl_xor(t, 1, 32));
            t = fmaxf(t, __shfl_xor(t, 2, 32));
            t = fmaxf(t, __shfl_xor(t, 4, 32));
            t = fmaxf(t, __shfl_xor(t, 8, 32));
            const float mnew  = fmaxf(mi[r], t);
            const float alpha = __expf(mi[r] - mnew);
            const float p0    = __expf(s0[r] - mnew);
            const float p1    = __expf(s1[r] - mnew);
            float rs = p0 + p1;
            rs += __shfl_xor(rs, 1, 32);
            rs += __shfl_xor(rs, 2, 32);
            rs += __shfl_xor(rs, 4, 32);
            rs += __shfl_xor(rs, 8, 32);
            li[r] = li[r] * alpha + rs;
            mi[r] = mnew;
            oacc[0][r] *= alpha; oacc[1][r] *= alpha;
            oacc[2][r] *= alpha; oacc[3][r] *= alpha;
            // C-layout -> LDS (row = M, col = key-in-tile), bf16
            ldsP[wave][half * 8 + r][l]      = bf_bits(p0);
            ldsP[wave][half * 8 + r][16 + l] = bf_bits(p1);
        }
        __syncthreads();

        // --- read P back in A-layout (16x32) ---
        v16bf ap = *(const v16bf*)&ldsP[wave][l][half * 16];

        // --- O += P * V ; B tiles = Vt columns d = t*16 + l ---
#pragma unroll
        for (int t = 0; t < 4; ++t) {
            const __bf16* vrow = Vp + (size_t)(t * 16 + l) * NTOK + kt + half * 16;
            v16bf bv = *(const v16bf*)vrow;
            oacc[t] = wmma_bf16(ap, bv, oacc[t]);
        }
        __syncthreads();
    }

    // --- normalize and write AO[b, n, head*64 + d] ---
#pragma unroll
    for (int r = 0; r < 8; ++r) {
        const float inv = li[r] > 0.0f ? 1.0f / li[r] : 0.0f;
        const int n = rt * 16 + half * 8 + r;
        const size_t base = ((size_t)(bb * NTOK + n)) * CDIM + head * HDIM;
#pragma unroll
        for (int t = 0; t < 4; ++t)
            AO[base + t * 16 + l] = f2bf(oacc[t][r] * inv);
    }
}

// ---------------------------------------------------------------------------
// Kernel 3: out = AO @ proj_w^T + proj_b (bf16 GEMM), f32 output.
// Grid: (288, 12) x 128 threads; ft = blockIdx.y*4 + wave in [0,48).
// ---------------------------------------------------------------------------
__global__ __launch_bounds__(128)
void proj_kernel(const __bf16* __restrict__ AO, const __bf16* __restrict__ w,
                 const float* __restrict__ bias, float* __restrict__ out)
{
    const int lane = threadIdx.x & 31;
    const int wave = threadIdx.x >> 5;
    const int l    = lane & 15;
    const int half = lane >> 4;
    const int mt   = blockIdx.x;
    const int ft   = blockIdx.y * 4 + wave;

    const __bf16* arow = AO + (size_t)(mt * 16 + l) * CDIM;
    const __bf16* wrow = w  + (size_t)(ft * 16 + l) * CDIM;

    v8f acc = {};
    for (int k0 = 0; k0 < CDIM; k0 += 32) {
        v16bf a = *(const v16bf*)(arow + k0 + half * 16);
        v16bf b = *(const v16bf*)(wrow + k0 + half * 16);
        acc = wmma_bf16(a, b, acc);
    }

    const int f = ft * 16 + l;
    const float bv = bias[f];
#pragma unroll
    for (int r = 0; r < 8; ++r) {
        const int token = mt * 16 + half * 8 + r;
        out[(size_t)token * CDIM + f] = acc[r] + bv;
    }
}

// ---------------------------------------------------------------------------
extern "C" void kernel_launch(void* const* d_in, const int* in_sizes, int n_in,
                              void* d_out, int out_size, void* d_ws, size_t ws_size,
                              hipStream_t stream)
{
    (void)in_sizes; (void)n_in; (void)out_size; (void)ws_size;

    const float* x      = (const float*)d_in[0];
    // d_in[1] = H, d_in[2] = W (unused; N fixed at 48*48)
    const float* qkv_w  = (const float*)d_in[3];
    const float* proj_w = (const float*)d_in[4];
    const float* proj_b = (const float*)d_in[5];
    float* out          = (float*)d_out;

    // Workspace layout (bf16 elements).  XB (x as bf16) is aliased with AO:
    // XB is consumed by qkv_kernel; AO is first written by attn_kernel.
    const size_t QK_ELEMS = (size_t)BATCH * NHEAD * NTOK * HDIM; // 3,538,944
    const size_t X_ELEMS  = (size_t)BATCH * NTOK * CDIM;         // 3,538,944
    const size_t WQ_ELEMS = (size_t)3 * CDIM * CDIM;             // 1,769,472
    const size_t WP_ELEMS = (size_t)CDIM * CDIM;                 //   589,824

    __bf16* Qs  = (__bf16*)d_ws;
    __bf16* Ks  = Qs + QK_ELEMS;
    __bf16* Vt  = Ks + QK_ELEMS;
    __bf16* XB  = Vt + QK_ELEMS;   // aliased: x_bf16, later AO
    __bf16* AO  = XB;
    __bf16* WQ  = XB + X_ELEMS;
    __bf16* WP  = WQ + WQ_ELEMS;
    // total: 16,515,072 bf16 = 33.0 MB

    // 0) one-time f32 -> bf16 conversions (memory-bound, ~35 MB traffic)
    cvt_kernel<<<(int)(X_ELEMS / 8 / 256), 256, 0, stream>>>(x, XB, (int)(X_ELEMS / 8));
    cvt_kernel<<<(int)(WQ_ELEMS / 8 / 256), 256, 0, stream>>>(qkv_w, WQ, (int)(WQ_ELEMS / 8));
    cvt_kernel<<<(int)(WP_ELEMS / 8 / 256), 256, 0, stream>>>(proj_w, WP, (int)(WP_ELEMS / 8));

    // 1) QKV projection: 288 x (144/4) blocks, 4 waves/block
    qkv_kernel<<<dim3(288, 36), 128, 0, stream>>>(XB, WQ, Qs, Ks, Vt);
    // 2) Flash attention: 36 x (B*H) blocks
    attn_kernel<<<dim3(36, 24), 128, 0, stream>>>(Qs, Ks, Vt, AO);
    // 3) Output projection: 288 x (48/4) blocks
    proj_kernel<<<dim3(288, 12), 128, 0, stream>>>(AO, WP, proj_b, out);
}
